// MultiSimilarityLoss_1769526526575
// MI455X (gfx1250) — matrix operand order repaired and checked by reading, not compiled, not run on hardware
//
#include <hip/hip_runtime.h>
#include <hip/hip_bf16.h>
#include <math.h>

// ---------------------------------------------------------------------------
// Multi-Similarity loss on MI455X (gfx1250).
// sim = Acol(8192x128) @ Arow(8192x128)^T computed twice with split-bf16
// (hi/lo) WMMA 16x16x32 accumulating in f32 (4 wmma per K-step ~ f32 accuracy).
// Pass1 mines per-row min_pos/max_neg with 2 M-tiles per wave (B-fragment
// loads amortized 2:1 over WMMAs, branchless min/max updates); launch_bounds
// (128,1) grants the full VGPR budget so A fragments stay resident.
// Pass2 accumulates masked sum-exps (1 M-tile per wave; exp-bound).
// Pass3 reduces to the scalar loss. Operands (8 MB) live in L2; the 256 MB
// sim matrix is never materialized.
// ---------------------------------------------------------------------------

#define DIM        128
#define THRESH_    0.5f
#define MARGIN_    0.1f
#define SCALE_POS_ 2.0f
#define SCALE_NEG_ 40.0f
#define EPS_       1e-5f
#define NSPLIT     8
#define WAVES1     4      // waves per block in pass1 (2 M-tiles each)
#define WAVES2     8      // waves per block in pass2 (1 M-tile each)

typedef __bf16 bf16_t;
typedef bf16_t v16bf __attribute__((ext_vector_type(16)));
typedef bf16_t v8bf  __attribute__((ext_vector_type(8)));
typedef float  v8f   __attribute__((ext_vector_type(8)));

// order-preserving float<->int map (involution); assumes no NaN inputs
__device__ __forceinline__ int   f2ord(float f) { int i = __float_as_int(f); return i >= 0 ? i : (i ^ 0x7FFFFFFF); }
__device__ __forceinline__ float ord2f(int i)   { return __int_as_float(i >= 0 ? i : (i ^ 0x7FFFFFFF)); }

// A fragment, 16x32 bf16 (MxK): lanes0-15 row=lane&15, halves0..7=K kb..kb+7,
// halves8..15=K kb+16..kb+23; lanes16-31 same rows, K offset +8.
__device__ __forceinline__ v16bf load_fragA(const bf16_t* __restrict__ base,
                                            int row, int kb, int laneHi) {
    const bf16_t* p = base + (size_t)row * DIM + kb + (laneHi ? 8 : 0);
    union { v16bf v; v8bf h[2]; } u;
    u.h[0] = *(const v8bf*)(p);
    u.h[1] = *(const v8bf*)(p + 16);
    return u.v;
}

// B fragment, 32x16 bf16 (KxN): lane holds column n=lane&15; lanes0-15
// halves0..7 = K kb..kb+7, halves8..15 = K kb+16..kb+23; lanes16-31 K offset +8.
__device__ __forceinline__ v16bf load_fragB(const bf16_t* __restrict__ base,
                                            int col, int kb, int laneHi) {
    const bf16_t* p = base + (size_t)col * DIM + kb + (laneHi ? 8 : 0);
    union { v16bf v; v8bf h[2]; } u;
    u.h[0] = *(const v8bf*)(p);
    u.h[1] = *(const v8bf*)(p + 16);
    return u.v;
}

__device__ __forceinline__ v8f wmma_acc(v16bf a, v16bf b, v8f c) {
    return __builtin_amdgcn_wmma_f32_16x16x32_bf16(false, a, false, b,
                                                   (short)0, c, false, false);
}

// ---------------------------------------------------------------------------
__global__ __launch_bounds__(256)
void msl_prepare(const float* __restrict__ colf,
                 const float* __restrict__ rowf,
                 bf16_t* __restrict__ Ahi, bf16_t* __restrict__ Alo,
                 bf16_t* __restrict__ Bhi, bf16_t* __restrict__ Blo,
                 int* __restrict__ minposE, int* __restrict__ maxnegE,
                 float* __restrict__ psum, float* __restrict__ nsum,
                 int nColElems, int nRowElems, int Bsz) {
    int idx = blockIdx.x * blockDim.x + threadIdx.x;
    if (idx < nColElems) {
        float x = colf[idx];
        bf16_t h = (bf16_t)x;
        Ahi[idx] = h;
        Alo[idx] = (bf16_t)(x - (float)h);
    } else if (idx < nColElems + nRowElems) {
        int j = idx - nColElems;
        float x = rowf[j];
        bf16_t h = (bf16_t)x;
        Bhi[j] = h;
        Blo[j] = (bf16_t)(x - (float)h);
    }
    if (idx < Bsz) {
        minposE[idx] = 0x7F800000;              // f2ord(+inf)
        maxnegE[idx] = (int)0x807FFFFF;         // f2ord(-inf)
        psum[idx] = 0.f;
        nsum[idx] = 0.f;
    }
}

// ---------------------------------------------------------------------------
// Pass 1: per-row min_pos / max_neg. 2 M-tiles (32 rows) per wave: B fragments
// are loaded once and feed both accumulators (2 WMMAs per b128 load).
// launch_bounds(128,1): register-fat by design; A fragments must stay in VGPRs.
__global__ __launch_bounds__(WAVES1 * 32, 1)
void msl_minmax(const bf16_t* __restrict__ Ahi, const bf16_t* __restrict__ Alo,
                const bf16_t* __restrict__ Bhi, const bf16_t* __restrict__ Blo,
                const int* __restrict__ tcol, const int* __restrict__ trow,
                int* __restrict__ minposE, int* __restrict__ maxnegE,
                int Bsz, int Nsz, int nChunk) {
    const int lane   = threadIdx.x & 31;
    const int laneHi = lane >> 4;
    const int ln     = lane & 15;
    const int mPair  = blockIdx.x * WAVES1 + (threadIdx.x >> 5);
    if (mPair * 32 >= Bsz) return;
    const int mBase  = mPair * 32;

    v16bf aH[2][DIM / 32], aL[2][DIM / 32];
#pragma unroll
    for (int t = 0; t < 2; ++t)
#pragma unroll
        for (int ks = 0; ks < DIM / 32; ++ks) {
            aH[t][ks] = load_fragA(Ahi, mBase + t * 16 + ln, ks * 32, laneHi);
            aL[t][ks] = load_fragA(Alo, mBase + t * 16 + ln, ks * 32, laneHi);
        }
    int tcv[2][8];
#pragma unroll
    for (int t = 0; t < 2; ++t)
#pragma unroll
        for (int r = 0; r < 8; ++r) tcv[t][r] = tcol[mBase + t * 16 + r + 8 * laneHi];

    float mp[2][8], mx[2][8];
#pragma unroll
    for (int t = 0; t < 2; ++t)
#pragma unroll
        for (int r = 0; r < 8; ++r) { mp[t][r] = INFINITY; mx[t][r] = -INFINITY; }

    const int n0 = blockIdx.y * nChunk;
    const int n1 = min(n0 + nChunk, Nsz);
    for (int nb = n0; nb < n1; nb += 16) {
        const int col = nb + ln;
        v8f acc0 = (v8f){0.f, 0.f, 0.f, 0.f, 0.f, 0.f, 0.f, 0.f};
        v8f acc1 = (v8f){0.f, 0.f, 0.f, 0.f, 0.f, 0.f, 0.f, 0.f};
#pragma unroll
        for (int ks = 0; ks < DIM / 32; ++ks) {
            // bh and bl consumed in separate phases -> only one live at a time
            {
                v16bf bh = load_fragB(Bhi, col, ks * 32, laneHi);
                acc0 = wmma_acc(aH[0][ks], bh, acc0);
                acc0 = wmma_acc(aL[0][ks], bh, acc0);
                acc1 = wmma_acc(aH[1][ks], bh, acc1);
                acc1 = wmma_acc(aL[1][ks], bh, acc1);
            }
            {
                v16bf bl = load_fragB(Blo, col, ks * 32, laneHi);
                acc0 = wmma_acc(aH[0][ks], bl, acc0);
                acc0 = wmma_acc(aL[0][ks], bl, acc0);
                acc1 = wmma_acc(aH[1][ks], bl, acc1);
                acc1 = wmma_acc(aL[1][ks], bl, acc1);
            }
        }
        const int tr = trow[col];
#pragma unroll
        for (int r = 0; r < 8; ++r) {
            {
                float s = acc0[r];
                bool same = (tr == tcv[0][r]);
                float posCand = (same && (s < 1.0f - EPS_)) ? s : INFINITY;
                float negCand = same ? -INFINITY : s;
                mp[0][r] = fminf(mp[0][r], posCand);
                mx[0][r] = fmaxf(mx[0][r], negCand);
            }
            {
                float s = acc1[r];
                bool same = (tr == tcv[1][r]);
                float posCand = (same && (s < 1.0f - EPS_)) ? s : INFINITY;
                float negCand = same ? -INFINITY : s;
                mp[1][r] = fminf(mp[1][r], posCand);
                mx[1][r] = fmaxf(mx[1][r], negCand);
            }
        }
    }
    // reduce across the 16 lanes that share each row (xor<16 stays in half-wave)
#pragma unroll
    for (int off = 8; off > 0; off >>= 1) {
#pragma unroll
        for (int t = 0; t < 2; ++t)
#pragma unroll
            for (int r = 0; r < 8; ++r) {
                mp[t][r] = fminf(mp[t][r], __shfl_xor(mp[t][r], off, 32));
                mx[t][r] = fmaxf(mx[t][r], __shfl_xor(mx[t][r], off, 32));
            }
    }
    if (ln == 0) {
#pragma unroll
        for (int t = 0; t < 2; ++t)
#pragma unroll
            for (int r = 0; r < 8; ++r) {
                int row = mBase + t * 16 + r + 8 * laneHi;
                atomicMin(&minposE[row], f2ord(mp[t][r]));
                atomicMax(&maxnegE[row], f2ord(mx[t][r]));
            }
    }
}

// ---------------------------------------------------------------------------
// Pass 2: masked sum-exp per row. 1 M-tile per wave (exp-bound loop; keep
// register pressure low for occupancy; branches let empty mask groups skip).
__global__ __launch_bounds__(WAVES2 * 32)
void msl_rowsum(const bf16_t* __restrict__ Ahi, const bf16_t* __restrict__ Alo,
                const bf16_t* __restrict__ Bhi, const bf16_t* __restrict__ Blo,
                const int* __restrict__ tcol, const int* __restrict__ trow,
                const int* __restrict__ minposE, const int* __restrict__ maxnegE,
                float* __restrict__ psum, float* __restrict__ nsum,
                int Bsz, int Nsz, int nChunk) {
    const int lane   = threadIdx.x & 31;
    const int laneHi = lane >> 4;
    const int ln     = lane & 15;
    const int mTile  = blockIdx.x * WAVES2 + (threadIdx.x >> 5);
    if (mTile * 16 >= Bsz) return;
    const int mBase  = mTile * 16;

    v16bf aH[DIM / 32], aL[DIM / 32];
#pragma unroll
    for (int ks = 0; ks < DIM / 32; ++ks) {
        aH[ks] = load_fragA(Ahi, mBase + ln, ks * 32, laneHi);
        aL[ks] = load_fragA(Alo, mBase + ln, ks * 32, laneHi);
    }
    int tcv[8]; float mpv[8], mxv[8];
#pragma unroll
    for (int r = 0; r < 8; ++r) {
        int row = mBase + r + 8 * laneHi;
        tcv[r] = tcol[row];
        mpv[r] = ord2f(minposE[row]);   // +inf if no positives
        mxv[r] = ord2f(maxnegE[row]);   // -inf if no negatives
    }
    float ps[8], ns[8];
#pragma unroll
    for (int r = 0; r < 8; ++r) { ps[r] = 0.f; ns[r] = 0.f; }

    const int n0 = blockIdx.y * nChunk;
    const int n1 = min(n0 + nChunk, Nsz);
    for (int nb = n0; nb < n1; nb += 16) {
        const int col = nb + ln;
        v8f acc = (v8f){0.f, 0.f, 0.f, 0.f, 0.f, 0.f, 0.f, 0.f};
#pragma unroll
        for (int ks = 0; ks < DIM / 32; ++ks) {
            {
                v16bf bh = load_fragB(Bhi, col, ks * 32, laneHi);
                acc = wmma_acc(aH[ks], bh, acc);
                acc = wmma_acc(aL[ks], bh, acc);
            }
            {
                v16bf bl = load_fragB(Blo, col, ks * 32, laneHi);
                acc = wmma_acc(aH[ks], bl, acc);
                acc = wmma_acc(aL[ks], bl, acc);
            }
        }
        const int tr = trow[col];
#pragma unroll
        for (int r = 0; r < 8; ++r) {
            float s = acc[r];
            if (tr == tcv[r]) {
                if (s < 1.0f - EPS_ && (s - MARGIN_) < mxv[r])
                    ps[r] += __expf(-SCALE_POS_ * (s - THRESH_));
            } else {
                if ((s + MARGIN_) > mpv[r])
                    ns[r] += __expf(SCALE_NEG_ * (s - THRESH_));
            }
        }
    }
#pragma unroll
    for (int off = 8; off > 0; off >>= 1) {
#pragma unroll
        for (int r = 0; r < 8; ++r) {
            ps[r] += __shfl_xor(ps[r], off, 32);
            ns[r] += __shfl_xor(ns[r], off, 32);
        }
    }
    if (ln == 0) {
#pragma unroll
        for (int r = 0; r < 8; ++r) {
            int row = mBase + r + 8 * laneHi;
            atomicAdd(&psum[row], ps[r]);
            atomicAdd(&nsum[row], ns[r]);
        }
    }
}

// ---------------------------------------------------------------------------
__global__ __launch_bounds__(256)
void msl_final(const float* __restrict__ psum, const float* __restrict__ nsum,
               float* __restrict__ out, int Bsz) {
    __shared__ float red[256];
    float acc = 0.f;
    for (int i = threadIdx.x; i < Bsz; i += 256) {
        float p = psum[i], n = nsum[i];
        if (p > 0.f && n > 0.f)
            acc += log1pf(p) / SCALE_POS_ + log1pf(n) / SCALE_NEG_;
    }
    red[threadIdx.x] = acc;
    __syncthreads();
    for (int s = 128; s > 0; s >>= 1) {
        if ((int)threadIdx.x < s) red[threadIdx.x] += red[threadIdx.x + s];
        __syncthreads();
    }
    if (threadIdx.x == 0) out[0] = red[0] / (float)Bsz;
}

// ---------------------------------------------------------------------------
extern "C" void kernel_launch(void* const* d_in, const int* in_sizes, int n_in,
                              void* d_out, int out_size, void* d_ws, size_t ws_size,
                              hipStream_t stream) {
    const float* inputs_col = (const float*)d_in[0];
    const int*   targets_col = (const int*)d_in[1];
    const float* inputs_row = (const float*)d_in[2];
    const int*   target_row = (const int*)d_in[3];

    const int Bsz = in_sizes[1];           // 8192
    const int Nsz = in_sizes[3];           // 8192
    const size_t nA = (size_t)Bsz * DIM;
    const size_t nB = (size_t)Nsz * DIM;

    char* w = (char*)d_ws;
    bf16_t* Ahi = (bf16_t*)w; w += nA * sizeof(bf16_t);
    bf16_t* Alo = (bf16_t*)w; w += nA * sizeof(bf16_t);
    bf16_t* Bhi = (bf16_t*)w; w += nB * sizeof(bf16_t);
    bf16_t* Blo = (bf16_t*)w; w += nB * sizeof(bf16_t);
    int*   minposE = (int*)w;   w += (size_t)Bsz * sizeof(int);
    int*   maxnegE = (int*)w;   w += (size_t)Bsz * sizeof(int);
    float* psum    = (float*)w; w += (size_t)Bsz * sizeof(float);
    float* nsum    = (float*)w;

    // pass 0: split to bf16 hi/lo + init accumulators
    {
        size_t total = nA + nB;
        int grid = (int)((total + 255) / 256);
        msl_prepare<<<grid, 256, 0, stream>>>(inputs_col, inputs_row,
                                              Ahi, Alo, Bhi, Blo,
                                              minposE, maxnegE, psum, nsum,
                                              (int)nA, (int)nB, Bsz);
    }
    const int nChunk = ((Nsz / NSPLIT + 15) / 16) * 16;

    // pass 1: per-row min_pos / max_neg mining stats (32 rows per wave)
    {
        const int mBlocks = (Bsz / 32 + WAVES1 - 1) / WAVES1;
        dim3 grid1(mBlocks, NSPLIT);
        msl_minmax<<<grid1, WAVES1 * 32, 0, stream>>>(Ahi, Alo, Bhi, Blo,
                                                      targets_col, target_row,
                                                      minposE, maxnegE, Bsz, Nsz, nChunk);
    }
    // pass 2: masked sum-exp per row (16 rows per wave)
    {
        const int mBlocks = (Bsz / 16 + WAVES2 - 1) / WAVES2;
        dim3 grid2(mBlocks, NSPLIT);
        msl_rowsum<<<grid2, WAVES2 * 32, 0, stream>>>(Ahi, Alo, Bhi, Blo,
                                                      targets_col, target_row,
                                                      minposE, maxnegE, psum, nsum,
                                                      Bsz, Nsz, nChunk);
    }
    // pass 3: scalar loss
    msl_final<<<1, 256, 0, stream>>>(psum, nsum, (float*)d_out, Bsz);
}
